// RelPosAttention_79809082294934
// MI455X (gfx1250) — matrix operand order repaired and checked by reading, not compiled, not run on hardware
//
#include <hip/hip_runtime.h>
#include <math.h>

#define DIM   1024
#define NQ    8192
#define NK    8192
#define HID   2048

typedef __attribute__((ext_vector_type(16))) __bf16 bf16x16;
typedef __attribute__((ext_vector_type(8)))  __bf16 bf16x8;
typedef __attribute__((ext_vector_type(4)))  __bf16 bf16x4;
typedef __attribute__((ext_vector_type(8)))  float  f32x8;
typedef __attribute__((ext_vector_type(4)))  float  f32x4;
typedef __attribute__((ext_vector_type(4)))  unsigned u32x4;
typedef __attribute__((ext_vector_type(8)))  unsigned u32x8;

union BF16x16U { bf16x16 v; bf16x8 h[2]; };

__device__ __forceinline__ f32x8 wmma_bf16(bf16x16 a, bf16x16 b, f32x8 c) {
  // D = A(16x32 bf16) * B(32x16 bf16) + C(16x16 f32)
  return __builtin_amdgcn_wmma_f32_16x16x32_bf16(false, a, false, b, (short)0, c,
                                                 false, false);
}

__device__ __forceinline__ f32x8 zero8() {
  f32x8 z;
#pragma unroll
  for (int i = 0; i < 8; ++i) z[i] = 0.f;
  return z;
}

__device__ __forceinline__ float fast_sqrt(float x) {
  return __builtin_amdgcn_sqrtf(x);   // v_sqrt_f32
}
__device__ __forceinline__ float fast_rcp(float x) {
  return __builtin_amdgcn_rcpf(x);    // v_rcp_f32
}

// ---- CDNA5 async global->LDS copy (ASYNCcnt), bypasses VGPR staging ----
__device__ __forceinline__ void async_ld16(void* lds_dst, const void* gsrc) {
  unsigned loff = (unsigned)(unsigned long long)lds_dst;  // LDS byte offset
  asm volatile("global_load_async_to_lds_b128 %0, %1, off"
               :
               : "v"(loff), "v"(gsrc)
               : "memory");
}
__device__ __forceinline__ void async_wait0() {
  asm volatile("s_wait_asynccnt 0x0" ::: "memory");
}

// ---- CDNA5 Tensor Data Mover: one instruction stages a 64 x 2048B tile ----
// D# per CDNA5 ISA: 2D tensor, data_size=8B (code 3),
// row = 256 units, tile = 256 x 64, contiguous (stride 256 units).
// LDS padding: pad_enable, pad_interval=7 (every 1024B), pad_amount=3 (16B)
// -> effective LDS row stride 2080B (2-way worst-case bank conflicts instead
//    of 16-way at a raw 2048B stride), all 16B chunks stay aligned.
#define TROW 2080   // bytes per padded row in LDS tile
#define TBLK 1040   // bytes per padded 1024B half-row block

__device__ __forceinline__ void tdm_load_tile(void* lds_dst, const void* gsrc) {
  unsigned long long ga = (unsigned long long)gsrc;
  u32x4 g0;
  g0[0] = 1u;                                            // count=1, user mode
  g0[1] = (unsigned)(unsigned long long)lds_dst;         // lds_addr
  g0[2] = (unsigned)ga;                                  // global_addr[31:0]
  g0[3] = ((unsigned)(ga >> 32) & 0x01FFFFFFu) | (2u << 30);  // [56:32] | type=2
  u32x8 g1;
  g1[0] = (3u << 16) | (1u << 20) | (7u << 22) | (3u << 25);
  //        data=8B     pad_en      ivl=1024B     amt=16B   (wg_mask=0)
  g1[1] = (256u << 16);   // tensor_dim0 = 256 units (low16)
  g1[2] = (8192u << 16);  // tensor_dim0 hi=0 | tensor_dim1 = 8192 (low16)
  g1[3] = (256u << 16);   // tensor_dim1 hi=0 | tile_dim0 = 256
  g1[4] = 64u;            // tile_dim1 = 64 | tile_dim2 = 0
  g1[5] = 256u;           // tensor_dim0_stride = 256 units
  g1[6] = 0u;
  g1[7] = 0u;
  asm volatile("tensor_load_to_lds %0, %1"
               :
               : "s"(g0), "s"(g1)
               : "memory");
}
__device__ __forceinline__ void tensor_wait0() {
  __builtin_amdgcn_s_wait_tensorcnt((unsigned short)0);
}

// ---- fragment loaders, plain row-major layout (ld in bf16 elements) ----
__device__ __forceinline__ bf16x16 load_a_frag(const __bf16* base, int row0, int k0,
                                               int ld, int l, int hi) {
  const __bf16* p = base + (size_t)(row0 + l) * ld + k0 + hi * 8;
  BF16x16U u;
  u.h[0] = *(const bf16x8*)(p);
  u.h[1] = *(const bf16x8*)(p + 16);
  return u.v;
}
__device__ __forceinline__ bf16x16 load_b_frag(const __bf16* base, int col0, int k0,
                                               int ld, int l, int hi) {
  const __bf16* p = base + (size_t)(col0 + l) * ld + k0 + hi * 8;
  BF16x16U u;
  u.h[0] = *(const bf16x8*)(p);
  u.h[1] = *(const bf16x8*)(p + 16);
  return u.v;
}

// ---- fragment loaders for the TDM-padded LDS tile layout ----
// element (r, k): byte = r*TROW + (k>>9)*TBLK + ((k&511) + sub)*2
// (k0 is a multiple of 32, so a 32-wide fragment never crosses the pad hole)
__device__ __forceinline__ bf16x16 load_a_frag_t(const char* base, int row0, int k0,
                                                 int l, int hi) {
  const char* p = base + (row0 + l) * TROW + (k0 >> 9) * TBLK
                  + ((k0 & 511) + hi * 8) * 2;
  BF16x16U u;
  u.h[0] = *(const bf16x8*)(p);
  u.h[1] = *(const bf16x8*)(p + 32);
  return u.v;
}
__device__ __forceinline__ bf16x16 load_b_frag_t(const char* base, int col0, int k0,
                                                 int l, int hi) {
  const char* p = base + (col0 + l) * TROW + (k0 >> 9) * TBLK
                  + ((k0 & 511) + hi * 8) * 2;
  BF16x16U u;
  u.h[0] = *(const bf16x8*)(p);
  u.h[1] = *(const bf16x8*)(p + 32);
  return u.v;
}

// dual-chain K-reduction over one 64-row tile (TDM layout):
// two independent accumulator chains -> two live fragment sets, ds-loads of one
// chain overlap the other chain's WMMA, and no back-to-back D->C WMMA stalls.
__device__ __forceinline__ f32x8 s_tile_dual(const char* Qs, const char* Ks,
                                             int qi, int ki, int l, int hi) {
  f32x8 acc0 = zero8(), acc1 = zero8();
#pragma unroll 4
  for (int kk = 0; kk < DIM; kk += 64) {
    bf16x16 a0 = load_a_frag_t(Qs, qi, kk, l, hi);
    bf16x16 b0 = load_b_frag_t(Ks, ki, kk, l, hi);
    bf16x16 a1 = load_a_frag_t(Qs, qi, kk + 32, l, hi);
    bf16x16 b1 = load_b_frag_t(Ks, ki, kk + 32, l, hi);
    acc0 = wmma_bf16(a0, b0, acc0);
    acc1 = wmma_bf16(a1, b1, acc1);
  }
  f32x8 acc;
#pragma unroll
  for (int i = 0; i < 8; ++i) acc[i] = acc0[i] + acc1[i];
  return acc;
}

// ---------------------------------------------------------------- small kernels

__global__ void w_transpose_k(const float* __restrict__ W, __bf16* __restrict__ WT,
                              int K, int N) {
  int idx = blockIdx.x * blockDim.x + threadIdx.x;
  if (idx >= K * N) return;
  int k = idx / N, n = idx % N;
  WT[(size_t)n * K + k] = (__bf16)W[idx];
}

__global__ void centers_k(const float* __restrict__ boxes, float* __restrict__ c, int n) {
  int i = blockIdx.x * blockDim.x + threadIdx.x;
  if (i >= n) return;
  const float* b = boxes + (size_t)i * 4;
  c[i * 2 + 0] = (b[0] + b[2]) * 0.5f;
  c[i * 2 + 1] = (b[1] + b[3]) * 0.5f;
}

__global__ void var_k(const float* __restrict__ kmat, float* __restrict__ gfv) {
  int c = blockIdx.x * blockDim.x + threadIdx.x;  // < DIM
  float s = 0.f, ss = 0.f;
  for (int r = 0; r < NK; ++r) {
    float x = kmat[(size_t)r * DIM + c];
    s += x; ss += x * x;
  }
  gfv[c] = (ss - s * s / (float)NK) / (float)(NK - 1);  // ddof=1
}

__global__ void mlp1_k(const float* __restrict__ gfv, const float* __restrict__ W1,
                       const float* __restrict__ b1, float* __restrict__ h1) {
  int j = blockIdx.x * blockDim.x + threadIdx.x;  // < HID
  float a = b1[j];
  for (int c = 0; c < DIM; ++c) a += gfv[c] * W1[(size_t)c * HID + j];
  h1[j] = fmaxf(a, 0.f);
}

__global__ void mlp2_k(const float* __restrict__ h1, const float* __restrict__ W2,
                       const float* __restrict__ b2, float* __restrict__ bw) {
  __shared__ float red[256];
  float p = 0.f;
  for (int j = threadIdx.x; j < HID; j += 256) p += h1[j] * W2[j];
  red[threadIdx.x] = p;
  __syncthreads();
  for (int s = 128; s; s >>= 1) {
    if (threadIdx.x < s) red[threadIdx.x] += red[threadIdx.x + s];
    __syncthreads();
  }
  if (threadIdx.x == 0) {
    float z = red[0] + b2[0];
    bw[0] = fast_rcp(1.f + __expf(-z));
  }
}

// ----------------------------------------------------- GEMM: C = alpha*(A * BT^T)
#define GSTRIDE 48

template <int OUTMODE>
__global__ __launch_bounds__(256)
void gemm_bf16_k(const float* __restrict__ A, const __bf16* __restrict__ BT,
                 void* __restrict__ Cout, const float* __restrict__ bias,
                 float alpha, int M, int N, int K) {
  __shared__ __bf16 As[128 * GSTRIDE];
  __shared__ __bf16 Bs[128 * GSTRIDE];

  const int tid = threadIdx.x;
  const int w = tid >> 5, lane = tid & 31, l = lane & 15, hi = lane >> 4;
  const int mbase = blockIdx.x * 128, nbase = blockIdx.y * 128;
  const int wm = (w & 3) * 32, wn = (w >> 2) * 64;

  f32x8 acc[2][4];
#pragma unroll
  for (int mi = 0; mi < 2; ++mi)
#pragma unroll
    for (int ni = 0; ni < 4; ++ni) acc[mi][ni] = zero8();

  const int lrow = tid >> 1;
  const int lhalf = (tid & 1) * 16;

  for (int kk = 0; kk < K; kk += 32) {
    // B tile: async global->LDS, no VGPR round-trip
    const __bf16* gb = BT + (size_t)(nbase + lrow) * K + kk + lhalf;
    __bf16* lb = Bs + lrow * GSTRIDE + lhalf;
    async_ld16(lb, gb);
    async_ld16(lb + 8, gb + 8);
    // A tile: f32 -> bf16 conversion through VGPRs
    const float* ga = A + (size_t)(mbase + lrow) * K + kk + lhalf;
    __bf16* la = As + lrow * GSTRIDE + lhalf;
#pragma unroll
    for (int i = 0; i < 16; i += 4) {
      f32x4 v4 = *(const f32x4*)(ga + i);
      bf16x4 b4;
      b4[0] = (__bf16)v4[0]; b4[1] = (__bf16)v4[1];
      b4[2] = (__bf16)v4[2]; b4[3] = (__bf16)v4[3];
      *(bf16x4*)(la + i) = b4;
    }
    if (kk + 32 < K) __builtin_prefetch(ga + 32, 0, 0);
    async_wait0();
    __syncthreads();

    bf16x16 af[2], bfr[4];
#pragma unroll
    for (int mi = 0; mi < 2; ++mi)
      af[mi] = load_a_frag(As, wm + mi * 16, 0, GSTRIDE, l, hi);
#pragma unroll
    for (int ni = 0; ni < 4; ++ni)
      bfr[ni] = load_b_frag(Bs, wn + ni * 16, 0, GSTRIDE, l, hi);
#pragma unroll
    for (int mi = 0; mi < 2; ++mi)
#pragma unroll
      for (int ni = 0; ni < 4; ++ni)
        acc[mi][ni] = wmma_bf16(af[mi], bfr[ni], acc[mi][ni]);
    __syncthreads();
  }

#pragma unroll
  for (int mi = 0; mi < 2; ++mi)
#pragma unroll
    for (int ni = 0; ni < 4; ++ni) {
      int col = nbase + wn + ni * 16 + l;
      int mrow = mbase + wm + mi * 16 + hi * 8;
      if (OUTMODE == 1) {
        bf16x8 pk;
#pragma unroll
        for (int vv = 0; vv < 8; ++vv) pk[vv] = (__bf16)(acc[mi][ni][vv] * alpha);
        *(bf16x8*)((__bf16*)Cout + (size_t)col * M + mrow) = pk;
      } else if (OUTMODE == 2) {
        float bv = bias[col];
#pragma unroll
        for (int vv = 0; vv < 8; ++vv)
          ((float*)Cout)[(size_t)(mrow + vv) * N + col] = acc[mi][ni][vv] * alpha + bv;
      } else {
#pragma unroll
        for (int vv = 0; vv < 8; ++vv)
          ((__bf16*)Cout)[(size_t)(mrow + vv) * N + col] =
              (__bf16)(acc[mi][ni][vv] * alpha);
      }
    }
}

// ------------------------------------------------------- attention pass A: row max

__global__ __launch_bounds__(512)
void attn_rowmax_k(const __bf16* __restrict__ qh, const __bf16* __restrict__ kh,
                   const float* __restrict__ cq, const float* __restrict__ cr,
                   const float* __restrict__ bwp, float* __restrict__ rowmax) {
  extern __shared__ char smem[];
  char* Qs = smem;               // 64 x TROW (TDM-padded)
  char* Ks = smem + 64 * TROW;   // 64 x TROW
  __shared__ float pmax[64][4];

  const int tid = threadIdx.x;
  const int w = tid >> 5, lane = tid & 31, l = lane & 15, hi = lane >> 4;
  const int qbase = blockIdx.x * 64;
  const float bw = *bwp;

  if (w == 0) tdm_load_tile(Qs, qh + (size_t)qbase * DIM);  // one TDM op

  const int qi = (w & 3) * 16, ki = (w >> 2) * 16;
  float cqx8[8], cqy8[8], runmax[8];
#pragma unroll
  for (int vv = 0; vv < 8; ++vv) {
    int rg = qbase + qi + hi * 8 + vv;
    cqx8[vv] = cq[rg * 2 + 0];
    cqy8[vv] = cq[rg * 2 + 1];
    runmax[vv] = -3.402823466e38f;
  }

  for (int kt = 0; kt < NK; kt += 64) {
    __syncthreads();
    if (w == 0) {
      tdm_load_tile(Ks, kh + (size_t)kt * DIM);
      tensor_wait0();             // covers Q-tile TDM too on first iteration
    }
    __syncthreads();

    f32x8 acc = s_tile_dual(Qs, Ks, qi, ki, l, hi);

    int cg = kt + ki + l;
    float crx = cr[cg * 2 + 0], cry = cr[cg * 2 + 1];
#pragma unroll
    for (int vv = 0; vv < 8; ++vv) {
      float dx = cqx8[vv] - crx, dy = cqy8[vv] - cry;
      float s = acc[vv] - bw * fast_sqrt(dx * dx + dy * dy + 1e-8f);
      runmax[vv] = fmaxf(runmax[vv], s);
    }
  }

#pragma unroll
  for (int vv = 0; vv < 8; ++vv)
    for (int off = 8; off; off >>= 1)
      runmax[vv] = fmaxf(runmax[vv], __shfl_xor(runmax[vv], off, 16));
  if (l == 0)
#pragma unroll
    for (int vv = 0; vv < 8; ++vv) pmax[qi + hi * 8 + vv][w >> 2] = runmax[vv];
  __syncthreads();
  if (tid < 64) {
    float m = fmaxf(fmaxf(pmax[tid][0], pmax[tid][1]),
                    fmaxf(pmax[tid][2], pmax[tid][3]));
    rowmax[qbase + tid] = m;
  }
}

// ---------------------------------- attention pass B: P = exp(S-m), x = (P*V)/l
#define PLD 72

__global__ __launch_bounds__(512)
void attn_pv_k(const __bf16* __restrict__ qh, const __bf16* __restrict__ kh,
               const __bf16* __restrict__ vT, const float* __restrict__ cq,
               const float* __restrict__ cr, const float* __restrict__ bwp,
               const float* __restrict__ rowmax, float* __restrict__ xout) {
  extern __shared__ char smem[];
  char* Qs = smem;
  char* Ks = smem + 64 * TROW;
  __bf16* Ps = (__bf16*)(smem + 2 * 64 * TROW);  // 64 x PLD
  __shared__ float psum_part[64][4];
  __shared__ float lsum[64];

  const int tid = threadIdx.x;
  const int w = tid >> 5, lane = tid & 31, l = lane & 15, hi = lane >> 4;
  const int qbase = blockIdx.x * 64;
  const float bw = *bwp;

  if (w == 0) tdm_load_tile(Qs, qh + (size_t)qbase * DIM);

  const int qi = (w & 3) * 16, ki = (w >> 2) * 16;  // S-phase mapping (4x4 waves)
  const int dbase = w * 64;                         // PV-phase: 64 d-rows per wave

  float cqx8[8], cqy8[8], m8[8], rs[8];
#pragma unroll
  for (int vv = 0; vv < 8; ++vv) {
    int rg = qbase + qi + hi * 8 + vv;
    cqx8[vv] = cq[rg * 2 + 0];
    cqy8[vv] = cq[rg * 2 + 1];
    m8[vv] = rowmax[rg];
    rs[vv] = 0.f;
  }

  f32x8 xacc[4][4];  // [d-frag][q-frag] -> xT block [64 d][64 q]
#pragma unroll
  for (int mf = 0; mf < 4; ++mf)
#pragma unroll
    for (int nf = 0; nf < 4; ++nf) xacc[mf][nf] = zero8();

  for (int kt = 0; kt < NK; kt += 64) {
    __syncthreads();  // previous PV done before Ks overwrite
    if (w == 0) {
      tdm_load_tile(Ks, kh + (size_t)kt * DIM);
      tensor_wait0();
    }
    if (kt + 64 < NK)
      __builtin_prefetch(vT + (size_t)(dbase + lane) * NK + kt + 64, 0, 0);
    __syncthreads();

    f32x8 acc = s_tile_dual(Qs, Ks, qi, ki, l, hi);

    int cg = kt + ki + l;
    float crx = cr[cg * 2 + 0], cry = cr[cg * 2 + 1];
#pragma unroll
    for (int vv = 0; vv < 8; ++vv) {
      float dx = cqx8[vv] - crx, dy = cqy8[vv] - cry;
      float s = acc[vv] - bw * fast_sqrt(dx * dx + dy * dy + 1e-8f);
      float p = __expf(s - m8[vv]);
      rs[vv] += p;
      Ps[(qi + hi * 8 + vv) * PLD + ki + l] = (__bf16)p;
    }
    __syncthreads();

    // PV: xT[64 d][64 q] += vT-rows * P^T (V frags pipelined one mf-step ahead)
#pragma unroll
    for (int ks = 0; ks < 64; ks += 32) {
      bf16x16 bq[4];
#pragma unroll
      for (int nf = 0; nf < 4; ++nf)
        bq[nf] = load_b_frag(Ps, nf * 16, ks, PLD, l, hi);
      bf16x16 av = load_a_frag(vT, dbase, kt + ks, NK, l, hi);  // global
#pragma unroll
      for (int mf = 0; mf < 4; ++mf) {
        bf16x16 avn;
        if (mf < 3)
          avn = load_a_frag(vT, dbase + (mf + 1) * 16, kt + ks, NK, l, hi);
#pragma unroll
        for (int nf = 0; nf < 4; ++nf)
          xacc[mf][nf] = wmma_bf16(av, bq[nf], xacc[mf][nf]);
        av = avn;
      }
    }
  }

  // finish row sums
#pragma unroll
  for (int vv = 0; vv < 8; ++vv)
    for (int off = 8; off; off >>= 1) rs[vv] += __shfl_xor(rs[vv], off, 16);
  if (l == 0)
#pragma unroll
    for (int vv = 0; vv < 8; ++vv) psum_part[qi + hi * 8 + vv][w >> 2] = rs[vv];
  __syncthreads();
  if (tid < 64)
    lsum[tid] = psum_part[tid][0] + psum_part[tid][1] +
                psum_part[tid][2] + psum_part[tid][3];
  __syncthreads();

  // store x[q][d] = xT/lsum (8 contiguous d per lane per frag)
#pragma unroll
  for (int mf = 0; mf < 4; ++mf)
#pragma unroll
    for (int nf = 0; nf < 4; ++nf) {
      int ql = nf * 16 + l;
      float inv = fast_rcp(lsum[ql]);
      int rowg = qbase + ql;
      int d0 = dbase + mf * 16 + hi * 8;
      float* dst = xout + (size_t)rowg * DIM + d0;
      f32x4 lo, hi4;
#pragma unroll
      for (int vv = 0; vv < 4; ++vv) lo[vv] = xacc[mf][nf][vv] * inv;
#pragma unroll
      for (int vv = 0; vv < 4; ++vv) hi4[vv] = xacc[mf][nf][vv + 4] * inv;
      *(f32x4*)(dst) = lo;
      *(f32x4*)(dst + 4) = hi4;
    }
}

// ---------------------------------------------------------------------- launch

extern "C" void kernel_launch(void* const* d_in, const int* in_sizes, int n_in,
                              void* d_out, int out_size, void* d_ws, size_t ws_size,
                              hipStream_t stream) {
  const float* q  = (const float*)d_in[0];
  const float* k  = (const float*)d_in[1];
  const float* v  = (const float*)d_in[2];
  const float* ref_boxes   = (const float*)d_in[3];
  const float* query_boxes = (const float*)d_in[4];
  const float* Wq = (const float*)d_in[5];
  const float* Wk = (const float*)d_in[6];
  const float* Wv = (const float*)d_in[7];
  const float* Wp = (const float*)d_in[8];
  const float* bp = (const float*)d_in[9];
  const float* W1 = (const float*)d_in[10];
  const float* b1 = (const float*)d_in[11];
  const float* W2 = (const float*)d_in[12];
  const float* b2 = (const float*)d_in[13];

  char* ws = (char*)d_ws;
  size_t o = 0;
  auto take = [&](size_t bytes) -> char* {
    char* p = ws + o;
    o += (bytes + 255) & ~(size_t)255;
    return p;
  };
  __bf16* WqT = (__bf16*)take((size_t)DIM * DIM * 2);
  __bf16* WkT = (__bf16*)take((size_t)DIM * DIM * 2);
  __bf16* WvT = (__bf16*)take((size_t)DIM * DIM * 2);
  __bf16* WpT = (__bf16*)take((size_t)DIM * DIM * 2);
  __bf16* qhB = (__bf16*)take((size_t)NQ * DIM * 2);
  __bf16* khB = (__bf16*)take((size_t)NK * DIM * 2);
  __bf16* vTB = (__bf16*)take((size_t)NK * DIM * 2);   // [DIM][NK]
  float*  xbuf = (float*)take((size_t)NQ * DIM * 4);
  float*  cqc = (float*)take((size_t)NQ * 2 * 4);
  float*  crc = (float*)take((size_t)NK * 2 * 4);
  float*  gfv = (float*)take((size_t)DIM * 4);
  float*  h1  = (float*)take((size_t)HID * 4);
  float*  bwv = (float*)take(256);
  float*  rmax = (float*)take((size_t)NQ * 4);

  dim3 tb(256);
  w_transpose_k<<<dim3(DIM * DIM / 256), tb, 0, stream>>>(Wq, WqT, DIM, DIM);
  w_transpose_k<<<dim3(DIM * DIM / 256), tb, 0, stream>>>(Wk, WkT, DIM, DIM);
  w_transpose_k<<<dim3(DIM * DIM / 256), tb, 0, stream>>>(Wv, WvT, DIM, DIM);
  w_transpose_k<<<dim3(DIM * DIM / 256), tb, 0, stream>>>(Wp, WpT, DIM, DIM);

  centers_k<<<dim3(NQ / 256), tb, 0, stream>>>(query_boxes, cqc, NQ);
  centers_k<<<dim3(NK / 256), tb, 0, stream>>>(ref_boxes, crc, NK);

  var_k<<<dim3(DIM / 256), tb, 0, stream>>>(k, gfv);
  mlp1_k<<<dim3(HID / 256), tb, 0, stream>>>(gfv, W1, b1, h1);
  mlp2_k<<<dim3(1), tb, 0, stream>>>(h1, W2, b2, bwv);

  const float scale = 0.03125f;  // 1024^-0.5
  dim3 gg(NQ / 128, DIM / 128);
  gemm_bf16_k<0><<<gg, tb, 0, stream>>>(q, WqT, qhB, nullptr, scale, NQ, DIM, DIM);
  gemm_bf16_k<0><<<gg, tb, 0, stream>>>(k, WkT, khB, nullptr, 1.0f, NK, DIM, DIM);
  gemm_bf16_k<1><<<gg, tb, 0, stream>>>(v, WvT, vTB, nullptr, 1.0f, NK, DIM, DIM);

  const size_t smemA = (size_t)2 * 64 * TROW;                 // Q + K TDM tiles
  const size_t smemB = smemA + (size_t)64 * PLD * 2;          // + P tile
  attn_rowmax_k<<<dim3(NQ / 64), dim3(512), smemA, stream>>>(qhB, khB, cqc, crc,
                                                             bwv, rmax);
  attn_pv_k<<<dim3(NQ / 64), dim3(512), smemB, stream>>>(qhB, khB, vTB, cqc, crc,
                                                         bwv, rmax, xbuf);

  gemm_bf16_k<2><<<gg, tb, 0, stream>>>(xbuf, WpT, d_out, bp, 1.0f, NQ, DIM, DIM);
}
